// ConcatMambaFusionBlock_77275051589817
// MI455X (gfx1250) — matrix operand order repaired and checked by reading, not compiled, not run on hardware
//
#include <hip/hip_runtime.h>
#include <hip/hip_bf16.h>

typedef __attribute__((ext_vector_type(16))) _Float16 v16h;
typedef __attribute__((ext_vector_type(8)))  float    v8f;

// Problem constants
#define DMODEL 96
#define DINNER 192
#define C2     384
#define LTOT   4096            // H*W = 64*64
#define MROWS  8192            // B*L
#define NSTATE 4
#define DTRANK 6
#define NCHUNK 16              // scan chunks
#define CLEN   256             // LTOT / NCHUNK

__device__ inline v8f v8f_zero() {
    v8f v;
#pragma unroll
    for (int i = 0; i < 8; ++i) v[i] = 0.0f;
    return v;
}

// Load a 16x32 f16 A-fragment from a row-major f32 matrix (ld floats per row).
// ISA layout (16-bit A 16x32): lane m=lane&15; K runs [kb..kb+7] and [kb+16..kb+23],
// kb = (lane>=16 ? 8 : 0), all offset by k0.
__device__ inline v16h load_a_frag_f32(const float* __restrict__ base, int ld,
                                       int row, int k0, int lane) {
    const int kb = k0 + ((lane >> 4) << 3);
    const float* p0 = base + (size_t)row * ld + kb;
    v16h a;
#pragma unroll
    for (int j = 0; j < 8; ++j) a[j] = (_Float16)p0[j];
#pragma unroll
    for (int j = 0; j < 8; ++j) a[8 + j] = (_Float16)p0[16 + j];
    return a;
}

// Load a 32x16 f16 B-fragment from LDS where weights are stored transposed:
// Wt[n*ldk + k] (halves). ISA layout: N = lane&15, K = k0 + 8*(lane>=16) + (j<8 ? j : j+8).
__device__ inline v16h load_b_frag_lds(const _Float16* __restrict__ Wt, int ldk,
                                       int nbase, int k0, int lane) {
    const _Float16* p = Wt + (size_t)(nbase + (lane & 15)) * ldk + k0 + ((lane >> 4) << 3);
    v16h b;
#pragma unroll
    for (int j = 0; j < 16; ++j) b[j] = p[j < 8 ? j : j + 8];
    return b;
}

__device__ inline float fast_silu(float x) {
    return x / (1.0f + __expf(-x));
}
__device__ inline float fast_softplus(float x) {
    return (x > 20.0f) ? x : __logf(1.0f + __expf(x));
}

// direction index remap (involution): scan position -> pixel index in xT
__device__ inline int dir_map(int k, int l) {
    if (k == 0) return l;
    if (k == 1) return ((l & 63) << 6) | (l >> 6);     // H/W transpose
    if (k == 2) return (LTOT - 1) - l;                  // flip
    const int lf = (LTOT - 1) - l;                      // flip of transpose
    return ((lf & 63) << 6) | (lf >> 6);
}

// ---------------------------------------------------------------------------
// Kernel A: input projections  pr = x_rgb @ W_in_x,  pe = x_e @ W_in_y  (WMMA)
// cols [0,192)  -> convin (B, 384, L)  channel = sel*192 + col   (NCHW for conv)
// cols [192,384)-> z      (B*L, 384)   channel = sel*192 + col-192
// grid (64, 3, 2), block 256 (8 waves). Each wave: fixed mtile, loops 8 n-tiles.
// ---------------------------------------------------------------------------
__global__ __launch_bounds__(256) void k_inproj(
    const float* __restrict__ x_rgb, const float* __restrict__ x_e,
    const float* __restrict__ Wx,    const float* __restrict__ Wy,
    float* __restrict__ convin, float* __restrict__ z)
{
    __shared__ _Float16 Wt[128 * 104];            // 128 cols x 96 k (padded), 26.6 KB
    const int sel = blockIdx.z;
    const float* __restrict__ X = sel ? x_e : x_rgb;
    const float* __restrict__ W = sel ? Wy : Wx;
    const int n0  = blockIdx.y * 128;
    const int tid = threadIdx.x;

    for (int idx = tid; idx < 96 * 128; idx += 256) {
        const int k = idx >> 7, n = idx & 127;    // coalesced over n
        Wt[n * 104 + k] = (_Float16)W[k * C2 + n0 + n];
    }
    __syncthreads();

    const int wave = tid >> 5, lane = tid & 31;
    const int mtile = blockIdx.x * 8 + wave;      // 0..511
    const int lh = lane >> 4, nlane = lane & 15;
    const int row = mtile * 16 + nlane;

    v16h a[3];
#pragma unroll
    for (int kc = 0; kc < 3; ++kc) a[kc] = load_a_frag_f32(X, DMODEL, row, kc * 32, lane);

    for (int nt = 0; nt < 8; ++nt) {
        v8f acc = v8f_zero();
#pragma unroll
        for (int kc = 0; kc < 3; ++kc) {
            v16h b = load_b_frag_lds(Wt, 104, nt * 16, kc * 32, lane);
            acc = __builtin_amdgcn_wmma_f32_16x16x32_f16(false, a[kc], false, b,
                                                         (short)0, acc, false, false);
        }
#pragma unroll
        for (int r = 0; r < 8; ++r) {
            const int gm = mtile * 16 + r + (lh << 3);
            const int gc = n0 + nt * 16 + nlane;
            const int bb = gm >> 12, l = gm & (LTOT - 1);
            const float v = acc[r];
            if (gc < DINNER) convin[((size_t)(bb * C2 + sel * DINNER + gc)) * LTOT + l] = v;
            else             z[(size_t)gm * C2 + sel * DINNER + (gc - DINNER)] = v;
        }
    }
}

// ---------------------------------------------------------------------------
// Kernel B: depthwise 3x3 conv (SAME, zero pad) + bias + SiLU.
// convin (B,384,L) -> xT (B, L, 384) channel-last for the scan/x_dbl consumers.
// ---------------------------------------------------------------------------
__global__ __launch_bounds__(256) void k_dwconv(
    const float* __restrict__ convin,
    const float* __restrict__ wx, const float* __restrict__ bx,
    const float* __restrict__ wy, const float* __restrict__ by,
    float* __restrict__ xT)
{
    const int gid = blockIdx.x * 256 + threadIdx.x;   // (b*384+ch)*4096 + l
    const int l  = gid & (LTOT - 1);
    const int bc = gid >> 12;
    const int ch = bc % C2;
    const int b  = bc / C2;
    const float* __restrict__ wsrc = (ch < DINNER) ? (wx + ch * 9) : (wy + (ch - DINNER) * 9);
    const float bias = (ch < DINNER) ? bx[ch] : by[ch - DINNER];
    const int h = l >> 6, w = l & 63;
    const float* __restrict__ inp = convin + (size_t)bc * LTOT;
    float s = bias;
#pragma unroll
    for (int dy = -1; dy <= 1; ++dy) {
        const int hh = h + dy;
        if ((unsigned)hh < 64u) {
#pragma unroll
            for (int dx = -1; dx <= 1; ++dx) {
                const int ww = w + dx;
                if ((unsigned)ww < 64u)
                    s += wsrc[(dy + 1) * 3 + (dx + 1)] * inp[hh * 64 + ww];
            }
        }
    }
    xT[(size_t)(b * LTOT + l) * C2 + ch] = fast_silu(s);
}

// ---------------------------------------------------------------------------
// Kernel C: x_dbl[b,k,l,0:14] = x_proj_weight[k] (14x384) . xs[b,k,:,l]
// xs is a remap of xT. Output layout [b,k,l,14] so the scan reads it contiguously.
// grid (16, 8), block 256.
// ---------------------------------------------------------------------------
__global__ __launch_bounds__(256) void k_xdbl(
    const float* __restrict__ xT, const float* __restrict__ xpw,
    float* __restrict__ xdbl)
{
    __shared__ float wl[14 * C2];                 // 21.5 KB
    const int bk = blockIdx.y;                    // b*4 + k
    const int k  = bk & 3, b = bk >> 2;
    const int tid = threadIdx.x;
    for (int idx = tid; idx < 14 * C2; idx += 256) wl[idx] = xpw[k * 14 * C2 + idx];
    __syncthreads();

    const int l  = blockIdx.x * 256 + tid;
    const int lm = dir_map(k, l);
    const float4* __restrict__ xr = (const float4*)(xT + (size_t)(b * LTOT + lm) * C2);

    float acc[14];
#pragma unroll
    for (int c = 0; c < 14; ++c) acc[c] = 0.0f;
    for (int d4 = 0; d4 < C2 / 4; ++d4) {
        const float4 xv = xr[d4];
        const int d = d4 * 4;
#pragma unroll
        for (int c = 0; c < 14; ++c) {
            acc[c] += wl[c * C2 + d] * xv.x + wl[c * C2 + d + 1] * xv.y +
                      wl[c * C2 + d + 2] * xv.z + wl[c * C2 + d + 3] * xv.w;
        }
    }
    float* __restrict__ outp = xdbl + (size_t)(bk * LTOT + l) * 14;
#pragma unroll
    for (int c = 0; c < 14; ++c) outp[c] = acc[c];
}

// ===========================================================================
// Chunked selective scan (3 passes). The recurrence s <- e*s + d is a linear
// map, so chunk-local results compose:  (P2,S2)o(P1,S1) = (P2*P1, P2*S1+S2).
// Pass 1: per-chunk scan from zero, record per-state product P and partial S.
// Pass 2: 16-step sequential prefix per (b,k,c,n) -> true chunk init states.
// Pass 3: replay each chunk from its init state, compute y, atomically add.
// Thread mapping for passes 1/3: idx = ((bk*16 + j)*384 + c); lanes are
// consecutive c so u-reads and atomics are coalesced 128B. 1536 waves total.
// ===========================================================================
__global__ __launch_bounds__(256) void k_scan_part(
    const float* __restrict__ xT, const float* __restrict__ xdbl,
    const float* __restrict__ dtw, const float* __restrict__ dtb,
    const float* __restrict__ A_logs,
    float* __restrict__ chunkP, float* __restrict__ chunkS)
{
    const int idx = blockIdx.x * 256 + threadIdx.x;   // 49152 total
    const int c  = idx % C2;
    const int tj = idx / C2;                          // bk*16 + j
    const int j  = tj & (NCHUNK - 1);
    const int bk = tj >> 4;
    const int k  = bk & 3, b = bk >> 2;
    const int kc = k * C2 + c;

    float w6[DTRANK];
#pragma unroll
    for (int r = 0; r < DTRANK; ++r) w6[r] = dtw[(size_t)kc * DTRANK + r];
    const float bias = dtb[kc];
    float Ar[NSTATE];
#pragma unroll
    for (int n = 0; n < NSTATE; ++n) Ar[n] = -__expf(A_logs[(size_t)kc * NSTATE + n]);

    const float* __restrict__ xd = xdbl + (size_t)bk * LTOT * 14;
    float s[NSTATE], P[NSTATE];
#pragma unroll
    for (int n = 0; n < NSTATE; ++n) { s[n] = 0.0f; P[n] = 1.0f; }

    const int l0 = j * CLEN;
    for (int t = 0; t < CLEN; ++t) {
        const int l = l0 + t;
        const float* __restrict__ p = xd + (size_t)l * 14;
        if (t + 32 < CLEN) __builtin_prefetch(p + 32 * 14, 0, 0);

        const float dt = p[0] * w6[0] + p[1] * w6[1] + p[2] * w6[2] +
                         p[3] * w6[3] + p[4] * w6[4] + p[5] * w6[5] + bias;
        const float delta = fast_softplus(dt);
        const int lm = dir_map(k, l);
        const float du = delta * xT[(size_t)(b * LTOT + lm) * C2 + c];
#pragma unroll
        for (int n = 0; n < NSTATE; ++n) {
            const float e = __expf(delta * Ar[n]);
            P[n] *= e;
            s[n] = s[n] * e + du * p[6 + n];
        }
    }
    const size_t o = ((size_t)tj * C2 + c) * NSTATE;
#pragma unroll
    for (int n = 0; n < NSTATE; ++n) { chunkP[o + n] = P[n]; chunkS[o + n] = s[n]; }
}

__global__ __launch_bounds__(256) void k_scan_prefix(
    const float* __restrict__ chunkP, const float* __restrict__ chunkS,
    float* __restrict__ chunkI)
{
    const int idx = blockIdx.x * 256 + threadIdx.x;   // 12288 = 3072*4
    const int n  = idx & 3;
    const int t  = idx >> 2;                          // bk*384 + c
    const int c  = t % C2;
    const int bk = t / C2;
    float s = 0.0f;
    for (int j = 0; j < NCHUNK; ++j) {
        const size_t o = ((size_t)(bk * NCHUNK + j) * C2 + c) * NSTATE + n;
        chunkI[o] = s;
        s = chunkP[o] * s + chunkS[o];
    }
}

__global__ __launch_bounds__(256) void k_scan_final(
    const float* __restrict__ xT, const float* __restrict__ xdbl,
    const float* __restrict__ dtw, const float* __restrict__ dtb,
    const float* __restrict__ A_logs, const float* __restrict__ Ds,
    const float* __restrict__ chunkI, float* __restrict__ ycomb)
{
    const int idx = blockIdx.x * 256 + threadIdx.x;   // 49152 total
    const int c  = idx % C2;
    const int tj = idx / C2;
    const int j  = tj & (NCHUNK - 1);
    const int bk = tj >> 4;
    const int k  = bk & 3, b = bk >> 2;
    const int kc = k * C2 + c;

    float w6[DTRANK];
#pragma unroll
    for (int r = 0; r < DTRANK; ++r) w6[r] = dtw[(size_t)kc * DTRANK + r];
    const float bias = dtb[kc];
    float Ar[NSTATE];
#pragma unroll
    for (int n = 0; n < NSTATE; ++n) Ar[n] = -__expf(A_logs[(size_t)kc * NSTATE + n]);
    const float Dv = Ds[kc];

    float s[NSTATE];
    {
        const size_t o = ((size_t)tj * C2 + c) * NSTATE;
#pragma unroll
        for (int n = 0; n < NSTATE; ++n) s[n] = chunkI[o + n];
    }

    const float* __restrict__ xd = xdbl + (size_t)bk * LTOT * 14;
    const int l0 = j * CLEN;
    for (int t = 0; t < CLEN; ++t) {
        const int l = l0 + t;
        const float* __restrict__ p = xd + (size_t)l * 14;
        if (t + 32 < CLEN) __builtin_prefetch(p + 32 * 14, 0, 0);

        const float dt = p[0] * w6[0] + p[1] * w6[1] + p[2] * w6[2] +
                         p[3] * w6[3] + p[4] * w6[4] + p[5] * w6[5] + bias;
        const float delta = fast_softplus(dt);
        const int lm = dir_map(k, l);
        const float u  = xT[(size_t)(b * LTOT + lm) * C2 + c];
        const float du = delta * u;

        float y = u * Dv;
#pragma unroll
        for (int n = 0; n < NSTATE; ++n) {
            const float e = __expf(delta * Ar[n]);
            s[n] = s[n] * e + du * p[6 + n];
            y += s[n] * p[10 + n];
        }
        atomicAdd(&ycomb[(size_t)(b * LTOT + lm) * C2 + c], y);
    }
}

// ---------------------------------------------------------------------------
// Kernel E: LayerNorm over C=384 + SiLU(z) gate.  ycomb(B,L,C), z(B*L,C) -> yg
// grid (8192), block 128 (3 channels/thread).
// ---------------------------------------------------------------------------
__global__ __launch_bounds__(128) void k_lngate(
    const float* __restrict__ ycomb, const float* __restrict__ zbuf,
    const float* __restrict__ gamma, const float* __restrict__ beta,
    float* __restrict__ yg)
{
    __shared__ float r1[128], r2[128];
    const int r = blockIdx.x, t = threadIdx.x;
    const float* __restrict__ row = ycomb + (size_t)r * C2;
    const float v0 = row[t], v1 = row[t + 128], v2 = row[t + 256];
    r1[t] = v0 + v1 + v2;
    r2[t] = v0 * v0 + v1 * v1 + v2 * v2;
    __syncthreads();
    for (int off = 64; off > 0; off >>= 1) {
        if (t < off) { r1[t] += r1[t + off]; r2[t] += r2[t + off]; }
        __syncthreads();
    }
    const float mean = r1[0] * (1.0f / C2);
    const float var  = r2[0] * (1.0f / C2) - mean * mean;
    const float rstd = rsqrtf(var + 1e-5f);
    const float* __restrict__ zr = zbuf + (size_t)r * C2;
    float* __restrict__ og = yg + (size_t)r * C2;
    const float vv[3] = { v0, v1, v2 };
#pragma unroll
    for (int i = 0; i < 3; ++i) {
        const int cc = t + i * 128;
        const float yn = (vv[i] - mean) * rstd * gamma[cc] + beta[cc];
        og[cc] = yn * fast_silu(zr[cc]);
    }
}

// ---------------------------------------------------------------------------
// Kernel F: output projection  out = yg (8192x384) @ W_out (384x96)  (WMMA)
// Two LDS phases of K=192 each (38.4 KB) to stay within static-LDS limits.
// grid (64), block 256: wave -> mtile, 6 n-tiles per wave, 12 k-chunks total.
// ---------------------------------------------------------------------------
__global__ __launch_bounds__(256) void k_outproj(
    const float* __restrict__ yg, const float* __restrict__ Wout,
    float* __restrict__ out)
{
    __shared__ _Float16 Wt[96 * 200];             // 96 cols x 192 k (padded), 38.4 KB
    const int tid = threadIdx.x;
    const int wave = tid >> 5, lane = tid & 31;
    const int mtile = blockIdx.x * 8 + wave;      // 0..511
    const int lh = lane >> 4, nlane = lane & 15;
    const int row = mtile * 16 + nlane;

    v8f acc[6];
#pragma unroll
    for (int nt = 0; nt < 6; ++nt) acc[nt] = v8f_zero();

    for (int phase = 0; phase < 2; ++phase) {
        __syncthreads();
        for (int idx = tid; idx < 192 * 96; idx += 256) {
            const int n = idx % 96, kk = idx / 96;            // coalesced over n
            Wt[n * 200 + kk] = (_Float16)Wout[(phase * 192 + kk) * DMODEL + n];
        }
        __syncthreads();

        v16h a[6];
#pragma unroll
        for (int kc = 0; kc < 6; ++kc)
            a[kc] = load_a_frag_f32(yg, C2, row, phase * 192 + kc * 32, lane);

        for (int nt = 0; nt < 6; ++nt) {
#pragma unroll
            for (int kc = 0; kc < 6; ++kc) {
                v16h b = load_b_frag_lds(Wt, 200, nt * 16, kc * 32, lane);
                acc[nt] = __builtin_amdgcn_wmma_f32_16x16x32_f16(false, a[kc], false, b,
                                                                 (short)0, acc[nt], false, false);
            }
        }
    }

#pragma unroll
    for (int nt = 0; nt < 6; ++nt) {
#pragma unroll
        for (int r = 0; r < 8; ++r) {
            const int gm = mtile * 16 + r + (lh << 3);
            const int gn = nt * 16 + nlane;
            out[(size_t)gm * DMODEL + gn] = acc[nt][r];
        }
    }
}

// ---------------------------------------------------------------------------
// Host-side launcher
// ---------------------------------------------------------------------------
extern "C" void kernel_launch(void* const* d_in, const int* in_sizes, int n_in,
                              void* d_out, int out_size, void* d_ws, size_t ws_size,
                              hipStream_t stream)
{
    (void)in_sizes; (void)n_in; (void)out_size; (void)ws_size;
    const float* x_rgb          = (const float*)d_in[0];
    const float* x_e            = (const float*)d_in[1];
    const float* W_in_x         = (const float*)d_in[2];
    const float* W_in_y         = (const float*)d_in[3];
    const float* conv_w_x       = (const float*)d_in[4];
    const float* conv_b_x       = (const float*)d_in[5];
    const float* conv_w_y       = (const float*)d_in[6];
    const float* conv_b_y       = (const float*)d_in[7];
    const float* x_proj_weight  = (const float*)d_in[8];
    const float* dt_projs_w     = (const float*)d_in[9];
    const float* dt_projs_b     = (const float*)d_in[10];
    const float* A_logs         = (const float*)d_in[11];
    const float* Ds             = (const float*)d_in[12];
    const float* ln_gamma       = (const float*)d_in[13];
    const float* ln_beta        = (const float*)d_in[14];
    const float* W_out          = (const float*)d_in[15];

    // Workspace layout (floats). Total = 40 MB. Aliases:
    //   xT reused as yg after the scan; convin reused as ycomb after the conv.
    constexpr size_t N_BLC = (size_t)2 * LTOT * C2;              // 3,145,728
    constexpr size_t N_XD  = (size_t)2 * 4 * LTOT * 14;          //   458,752
    constexpr size_t N_CH  = (size_t)2 * 4 * NCHUNK * C2 * NSTATE; // 196,608
    float* ws     = (float*)d_ws;
    float* xT     = ws;                                  // (B, L, C)
    float* z      = ws + N_BLC;                          // (B*L, C)
    float* xdbl   = ws + 2 * N_BLC;                      // (B,K,L,14)
    float* convin = ws + 2 * N_BLC + N_XD;               // (B, C, L)
    float* chunkP = ws + 3 * N_BLC + N_XD;               // (bk, j, c, n)
    float* chunkS = chunkP + N_CH;
    float* chunkI = chunkS + N_CH;
    float* ycomb  = convin;                              // alias (conv consumed)
    float* yg     = xT;                                  // alias (xT consumed)

    k_inproj    <<<dim3(64, 3, 2), 256, 0, stream>>>(x_rgb, x_e, W_in_x, W_in_y, convin, z);
    k_dwconv    <<<dim3(12288),    256, 0, stream>>>(convin, conv_w_x, conv_b_x,
                                                     conv_w_y, conv_b_y, xT);
    k_xdbl      <<<dim3(16, 8),    256, 0, stream>>>(xT, x_proj_weight, xdbl);
    hipMemsetAsync(ycomb, 0, N_BLC * sizeof(float), stream);
    k_scan_part  <<<dim3(192), 256, 0, stream>>>(xT, xdbl, dt_projs_w, dt_projs_b,
                                                 A_logs, chunkP, chunkS);
    k_scan_prefix<<<dim3(48),  256, 0, stream>>>(chunkP, chunkS, chunkI);
    k_scan_final <<<dim3(192), 256, 0, stream>>>(xT, xdbl, dt_projs_w, dt_projs_b,
                                                 A_logs, Ds, chunkI, ycomb);
    k_lngate    <<<dim3(MROWS),    128, 0, stream>>>(ycomb, z, ln_gamma, ln_beta, yg);
    k_outproj   <<<dim3(64),       256, 0, stream>>>(yg, W_out, (float*)d_out);
}
